// Block_3599182594934
// MI455X (gfx1250) — matrix operand order repaired and checked by reading, not compiled, not run on hardware
//
#include <hip/hip_runtime.h>
#include <math.h>

// ---------------- problem constants ----------------
#define CH      384
#define LDQKV   1152
#define NH      6
#define HD      64
#define BATCH   32
#define T_TOK   513
#define TOK     16416         // 32*513
#define G_NUM   512
#define S_NUM   128
#define GS_     32
#define CHUNK_ROWS 16384
#define NCHUNK  8
#define CHUNK_GRP 512
#define HIDDEN  1536
#define REDD    16
#define PP      10
#define SCALE_ATT 0.125f

typedef __bf16 bf16_t;
typedef bf16_t v16bf __attribute__((ext_vector_type(16)));
typedef float  v8f   __attribute__((ext_vector_type(8)));
typedef int    v4i   __attribute__((ext_vector_type(4)));

__device__ __forceinline__ unsigned short f2bf(float f) {
  unsigned u = __float_as_uint(f);
  u += 0x7FFFu + ((u >> 16) & 1u);
  return (unsigned short)(u >> 16);
}
__device__ __forceinline__ float gelu_f(float x) {
  return 0.5f * x * (1.0f + erff(x * 0.70710678118654752f));
}

// ---------------- async global->LDS (CDNA5) with safe fallbacks ----------------
__device__ __forceinline__ void g2l_b128(const float* g, float* l) {
#if __has_builtin(__builtin_amdgcn_global_load_async_to_lds_b128)
  __builtin_amdgcn_global_load_async_to_lds_b128(
      (__attribute__((address_space(1))) v4i*)g,
      (__attribute__((address_space(3))) v4i*)l, 0, 0);
#elif defined(__gfx1250__)
  unsigned lo = (unsigned)(__UINTPTR_TYPE__)(__attribute__((address_space(3))) void*)l;
  asm volatile("global_load_async_to_lds_b128 %0, %1, off"
               :: "v"(lo), "v"((__attribute__((address_space(1))) v4i*)g)
               : "memory");
#else
  *(float4*)l = *(const float4*)g;
#endif
}
__device__ __forceinline__ void g2l_wait() {
#if __has_builtin(__builtin_amdgcn_s_wait_asynccnt)
  __builtin_amdgcn_s_wait_asynccnt(0);
#elif defined(__gfx1250__)
  asm volatile("s_wait_asynccnt 0x0" ::: "memory");
#endif
}

#define GF_GELU  1
#define GF_ACCUM 2

// ============ fp32 -> bf16 pack (4 at a time) ============
__global__ __launch_bounds__(256) void cvt_bf16_kernel(
    const float* __restrict__ s, unsigned short* __restrict__ d, long n4)
{
  long i = (long)blockIdx.x * blockDim.x + threadIdx.x;
  if (i >= n4) return;
  float4 v = ((const float4*)s)[i];
  uint2 o;
  o.x = (unsigned)f2bf(v.x) | ((unsigned)f2bf(v.y) << 16);
  o.y = (unsigned)f2bf(v.z) | ((unsigned)f2bf(v.w) << 16);
  ((uint2*)d)[i] = o;
}

// ============ batched GEMM: C = [accum +] alpha*act(A@B + bias) ============
// A: MxK bf16 row-major, B: KxN bf16 row-major, C: MxN fp32.
// 64x128 tile, 256 threads = 8 waves; wave -> 16x64 (4 accumulators).
// Requires K%4==0, N%4==0 (true for all uses). Branch-free clamped edge loads.
__global__ __launch_bounds__(256) void gemm_bf16(
    const unsigned short* __restrict__ A, const unsigned short* __restrict__ B,
    float* __restrict__ C, int M, int N, int K, long sA, long sB, long sC,
    const float* __restrict__ bias, int flags, float alpha, const int* __restrict__ alphaI)
{
  __shared__ unsigned short As[64][34];    // [m][k], stride 34 (17 banks)
  __shared__ unsigned short Bs[128][34];   // [n][k] transposed
  A += (long)blockIdx.z * sA;
  B += (long)blockIdx.z * sB;
  C += (long)blockIdx.z * sC;
  const int bm = blockIdx.y * 64, bn = blockIdx.x * 128;
  const int tid = threadIdx.x;
  const int w = tid >> 5, lane = tid & 31;
  const int wr = w & 3, wc = w >> 2;        // wave: rows [wr*16,+16), cols [wc*64,+64)
  const int hi = lane >> 4, lm = lane & 15;

  v8f acc[4];
  #pragma unroll
  for (int t = 0; t < 4; ++t) acc[t] = (v8f){0,0,0,0,0,0,0,0};

  for (int k0 = 0; k0 < K; k0 += 32) {
    // ---- A tile 64x32 : 512 quads, 2 per thread ----
    #pragma unroll
    for (int it = 0; it < 2; ++it) {
      int q = tid + it * 256;
      int r = q >> 3, kq = (q & 7) << 2;
      int gr = bm + r; gr = gr < M ? gr : M - 1;
      int gk = k0 + kq;
      int ok = gk < K;                       // K%4==0 -> whole quad valid or not
      int gkc = ok ? gk : 0;
      uint2 u = *(const uint2*)(A + (long)gr * K + gkc);
      unsigned msk = ok ? 0xFFFFFFFFu : 0u;
      *(unsigned*)&As[r][kq]     = u.x & msk;
      *(unsigned*)&As[r][kq + 2] = u.y & msk;
    }
    // ---- B tile 32x128 -> Bs[n][k] : 1024 quads, 4 per thread ----
    #pragma unroll
    for (int it = 0; it < 4; ++it) {
      int q = tid + it * 256;
      int k = q >> 5, n4 = (q & 31) << 2;
      int gk = k0 + k; gk = gk < K ? gk : K - 1;        // tail garbage killed by A mask
      int gn = bn + n4; gn = gn <= N - 4 ? gn : N - 4;  // dup reads, distinct LDS rows
      uint2 u = *(const uint2*)(B + (long)gk * N + gn);
      Bs[n4 + 0][k] = (unsigned short)(u.x);
      Bs[n4 + 1][k] = (unsigned short)(u.x >> 16);
      Bs[n4 + 2][k] = (unsigned short)(u.y);
      Bs[n4 + 3][k] = (unsigned short)(u.y >> 16);
    }
    __syncthreads();

    union Frag { v16bf v; unsigned u[8]; };
    Frag af, bf[4];
    const int arow = wr * 16 + lm;
    #pragma unroll
    for (int p = 0; p < 8; ++p) {
      // A 16x32 bf16: K(elem e) = (e>>3)*16 + hi*8 + (e&7); pair p -> elems 2p,2p+1
      int kk = ((p >> 2) << 4) + hi * 8 + ((p & 3) << 1);
      af.u[p] = *(const unsigned*)&As[arow][kk];
    }
    #pragma unroll
    for (int t = 0; t < 4; ++t) {
      const int bc = wc * 64 + t * 16 + lm;
      #pragma unroll
      for (int p = 0; p < 8; ++p) {
        int kk = ((p >> 2) << 4) + hi * 8 + ((p & 3) << 1);        // B 32x16 bf16: K(elem e) = (e>>3)*16 + hi*8 + (e&7)
        bf[t].u[p] = *(const unsigned*)&Bs[bc][kk];
      }
    }
    #pragma unroll
    for (int t = 0; t < 4; ++t)
      acc[t] = __builtin_amdgcn_wmma_f32_16x16x32_bf16(false, af.v, false, bf[t].v,
                                                       (short)0, acc[t], false, false);
    __syncthreads();
  }

  float a = alpha;
  if (alphaI) a *= (float)(*alphaI);
  #pragma unroll
  for (int e = 0; e < 8; ++e) {
    int row = bm + wr * 16 + e + 8 * hi;      // D: M = v + 8*hi, N = lane&15
    #pragma unroll
    for (int t = 0; t < 4; ++t) {
      int col = bn + wc * 64 + t * 16 + lm;
      if (row < M && col < N) {
        float val = acc[t][e];
        if (bias) val += bias[col];
        if (flags & GF_GELU) val = gelu_f(val);
        val *= a;
        long idx = (long)row * N + col;
        if (flags & GF_ACCUM) val += C[idx];
        C[idx] = val;
      }
    }
  }
}

// ============ attention with async LDS staging ============
__global__ __launch_bounds__(256) void attn_kernel(
    const float* __restrict__ qkv, float* __restrict__ out,
    int rowsPerGroup, int nq, int nkv, int ldout,
    const float* __restrict__ pqkv, int npk, const float* __restrict__ gate, float scale)
{
  __shared__ float Qs[16][68], Ks[16][68], Vs[16][68], Ps[16][16];
  const int g = blockIdx.x, h = blockIdx.y, qt = blockIdx.z;
  const int tid = threadIdx.x;
  const int qr = tid >> 4, kc = tid & 15;
  const long base = (long)g * rowsPerGroup;
  const int q0 = qt * 16;

  { // Q tile: 256 quads exactly (rows clamped; garbage rows never written out)
    int r = tid >> 4, d4 = (tid & 15) << 2;
    int row = q0 + r; row = row < nq ? row : nq - 1;
    g2l_b128(&qkv[(base + row) * LDQKV + h * HD + d4], &Qs[r][d4]);
  }

  float m = -1e30f, l = 0.0f, O[4] = {0, 0, 0, 0};
  for (int j0 = 0; j0 < nkv; j0 += 16) {
    __syncthreads();                          // previous tile fully consumed
    {
      int r = tid >> 4, d4 = (tid & 15) << 2;
      int row = j0 + r; row = row < nkv ? row : nkv - 1;   // masked via score
      g2l_b128(&qkv[(base + row) * LDQKV + CH + h * HD + d4],     &Ks[r][d4]);
      g2l_b128(&qkv[(base + row) * LDQKV + 2 * CH + h * HD + d4], &Vs[r][d4]);
    }
    g2l_wait();
    __syncthreads();
    float s = 0.0f;
    #pragma unroll 16
    for (int d = 0; d < 64; ++d) s += Qs[qr][d] * Ks[kc][d];
    s *= scale;
    if (j0 + kc >= nkv) s = -1e30f;
    float mx = s;
    #pragma unroll
    for (int off = 8; off; off >>= 1) mx = fmaxf(mx, __shfl_xor(mx, off, 16));
    float mnew = fmaxf(m, mx);
    float p = __expf(s - mnew);
    float ps = p;
    #pragma unroll
    for (int off = 8; off; off >>= 1) ps += __shfl_xor(ps, off, 16);
    float corr = __expf(m - mnew);
    l = l * corr + ps;
    m = mnew;
    Ps[qr][kc] = p;
    #pragma unroll
    for (int t = 0; t < 4; ++t) O[t] *= corr;
    __syncthreads();
    #pragma unroll 16
    for (int j = 0; j < 16; ++j) {
      float pj = Ps[qr][j];
      #pragma unroll
      for (int t = 0; t < 4; ++t) O[t] += pj * Vs[j][kc * 4 + t];
    }
  }
  float inv = (l > 0.0f) ? 1.0f / l : 0.0f;
  #pragma unroll
  for (int t = 0; t < 4; ++t) O[t] *= inv;

  if (pqkv) {   // gated prompt attention (separate softmax, npk keys)
    __syncthreads();
    {
      int r = tid >> 4, d4 = (tid & 15) << 2;
      int row = r < npk ? r : npk - 1;
      g2l_b128(&pqkv[(long)row * LDQKV + CH + h * HD + d4],     &Ks[r][d4]);
      g2l_b128(&pqkv[(long)row * LDQKV + 2 * CH + h * HD + d4], &Vs[r][d4]);
    }
    g2l_wait();
    __syncthreads();
    float s = 0.0f;
    #pragma unroll 16
    for (int d = 0; d < 64; ++d) s += Qs[qr][d] * Ks[kc][d];
    s *= scale;
    if (kc >= npk) s = -1e30f;
    float mx = s;
    #pragma unroll
    for (int off = 8; off; off >>= 1) mx = fmaxf(mx, __shfl_xor(mx, off, 16));
    float p = __expf(s - mx);
    float ps = p;
    #pragma unroll
    for (int off = 8; off; off >>= 1) ps += __shfl_xor(ps, off, 16);
    Ps[qr][kc] = p;
    __syncthreads();
    float P4[4] = {0, 0, 0, 0};
    #pragma unroll 16
    for (int j = 0; j < 16; ++j) {
      float pj = Ps[qr][j];
      #pragma unroll
      for (int t = 0; t < 4; ++t) P4[t] += pj * Vs[j][kc * 4 + t];
    }
    float gh = gate[h] / ps;
    #pragma unroll
    for (int t = 0; t < 4; ++t) O[t] += gh * P4[t];
  }

  int row = q0 + qr;
  if (row < nq) {
    #pragma unroll
    for (int t = 0; t < 4; ++t)
      out[(base + row) * ldout + h * HD + kc * 4 + t] = O[t];
  }
}

// ============ layernorm over C, one block per row ============
__global__ __launch_bounds__(128) void ln_kernel(
    const float* __restrict__ x, const float* __restrict__ g,
    const float* __restrict__ b, float* __restrict__ o, int Cc)
{
  __shared__ float r1[128], r2[128];
  long row = blockIdx.x;
  const float* xr = x + row * Cc;
  float s = 0.0f, s2 = 0.0f;
  for (int c = threadIdx.x; c < Cc; c += 128) { float v = xr[c]; s += v; s2 += v * v; }
  r1[threadIdx.x] = s; r2[threadIdx.x] = s2; __syncthreads();
  for (int off = 64; off; off >>= 1) {
    if ((int)threadIdx.x < off) { r1[threadIdx.x] += r1[threadIdx.x + off];
                                  r2[threadIdx.x] += r2[threadIdx.x + off]; }
    __syncthreads();
  }
  float mean = r1[0] / Cc;
  float var  = r2[0] / Cc - mean * mean;
  float inv  = rsqrtf(var + 1e-5f);
  for (int c = threadIdx.x; c < Cc; c += 128)
    o[row * Cc + c] = (xr[c] - mean) * inv * g[c] + b[c];
}

// ============ gather rows (float4) ============
__global__ __launch_bounds__(128) void gather_rows_kernel(
    const float* __restrict__ src, float* __restrict__ dst, const int* __restrict__ idx)
{
  long r = blockIdx.x;
  int v = idx[r];
  long srow = (long)(v / G_NUM) * T_TOK + 1 + (v % G_NUM);
  const float4* s4 = (const float4*)(src + srow * CH);
  float4* d4 = (float4*)(dst + r * CH);
  for (int c = threadIdx.x; c < CH / 4; c += 128) d4[c] = s4[c];
}

// ============ per-group max + mean over GS rows ============
__global__ __launch_bounds__(128) void group_reduce_kernel(
    const float* __restrict__ xnb, float* __restrict__ mx, float* __restrict__ mn,
    int Cc, int outBase)
{
  long grp = blockIdx.x;
  const float* p = xnb + grp * GS_ * Cc;
  for (int c = threadIdx.x; c < Cc; c += 128) {
    float vmax = -1e30f, vsum = 0.0f;
    #pragma unroll 8
    for (int r = 0; r < GS_; ++r) { float v = p[(long)r * Cc + c]; vmax = fmaxf(vmax, v); vsum += v; }
    mx[(outBase + grp) * Cc + c] = vmax;
    mn[(outBase + grp) * Cc + c] = vsum * (1.0f / GS_);
  }
}

// ============ vis = gelu(BN(max [+ mean])) + center_cof * x_ct ============
__global__ __launch_bounds__(256) void vis_kernel(
    const float* __restrict__ mx, const float* __restrict__ mn,
    const float* __restrict__ bng, const float* __restrict__ bnb,
    const float* __restrict__ bnm, const float* __restrict__ bnv,
    const float* __restrict__ xbuf, const int* __restrict__ cidx,
    const int* __restrict__ ifmm, const int* __restrict__ ccof,
    float* __restrict__ vis, long total, int Cc)
{
  long i = (long)blockIdx.x * blockDim.x + threadIdx.x;
  if (i >= total) return;
  long row = i / Cc; int c = (int)(i % Cc);
  float lc = mx[i];
  if (*ifmm) lc += mn[i];
  float t = (lc - bnm[c]) * rsqrtf(bnv[c] + 1e-5f) * bng[c] + bnb[c];
  t = gelu_f(t);
  int v = cidx[row];
  long srow = (long)(v / G_NUM) * T_TOK + 1 + (v % G_NUM);
  vis[i] = t + (float)(*ccof) * xbuf[srow * Cc + c];
}

// ============ inverse-distance weights, normalized over S ============
__global__ __launch_bounds__(128) void wgt_kernel(
    const float* __restrict__ c1, const float* __restrict__ c2, float* __restrict__ w)
{
  __shared__ float red[128];
  long bg = blockIdx.x;
  int b = (int)(bg / G_NUM);
  int s = threadIdx.x;
  float d = 0.0f;
  #pragma unroll
  for (int k = 0; k < 3; ++k) {
    float diff = c1[bg * 3 + k] - c2[((long)b * S_NUM + s) * 3 + k];
    d += diff * diff;
  }
  float wv = 1.0f / (d + 1e-8f);
  red[s] = wv; __syncthreads();
  for (int off = 64; off; off >>= 1) {
    if (s < off) red[s] += red[s + off];
    __syncthreads();
  }
  w[bg * S_NUM + s] = wv / red[0];
}

// ============ y = (*a)*x + y (float4) ============
__global__ __launch_bounds__(256) void axpy_kernel(
    float* __restrict__ y, const float* __restrict__ x, const float* __restrict__ ap, long n4)
{
  long i = (long)blockIdx.x * blockDim.x + threadIdx.x;
  if (i >= n4) return;
  float a = *ap;
  float4 xv = ((const float4*)x)[i];
  float4 yv = ((float4*)y)[i];
  yv.x += a * xv.x; yv.y += a * xv.y; yv.z += a * xv.z; yv.w += a * xv.w;
  ((float4*)y)[i] = yv;
}

// ---------------- workspace layout (float offsets) ----------------
#define F_XBUF  0L
#define F_XN    6303744L
#define F_QKV   12607488L
#define F_ATTN  31518720L
#define F_HBIG  37822464L
#define F_HBUF  63037440L
#define F_ADMID 69341184L
#define F_PQKV  69603840L
#define F_BW    69615360L      // bf16 weights (ushort region)
#define F_BACT  70813440L      // bf16 activations (up to 25.2M ushorts)
#define F_BACT2 83420928L      // bf16 adapter mid
// phase-2 reuse:
#define F_XNB    F_XN
#define F_LNNB   F_ATTN
#define F_QKVNB  F_QKV
#define F_ATTNNB F_HBUF
#define F_LCMAX  F_HBIG
#define F_LCMEAN (F_HBIG + 1572864L)
#define F_VIS    (F_HBIG + 3145728L)
#define F_WBUF   (F_HBIG + 4718592L)
// bf16 weight sub-offsets (ushort units)
#define oWQ  0L
#define oWP  442368L
#define oW1  589824L
#define oW2  1179648L
#define oAD  1769472L
#define oAU  1775616L
#define oA1Q 1781760L
#define oA1P 2224128L
#define oA1D 2371584L
#define oA1U 2377728L
#define oPE  2383872L

extern "C" void kernel_launch(void* const* d_in, const int* in_sizes, int n_in,
                              void* d_out, int out_size, void* d_ws, size_t ws_size,
                              hipStream_t stream)
{
  const float* x     = (const float*)d_in[0];
  const float* c1    = (const float*)d_in[1];
  const float* c2    = (const float*)d_in[2];
  const float* pe    = (const float*)d_in[4];
  const float* ln1g  = (const float*)d_in[5];
  const float* ln1b  = (const float*)d_in[6];
  const float* Wqkv  = (const float*)d_in[7];
  const float* Wproj = (const float*)d_in[8];
  const float* bproj = (const float*)d_in[9];
  const float* gate  = (const float*)d_in[10];
  const float* ln2g  = (const float*)d_in[11];
  const float* ln2b  = (const float*)d_in[12];
  const float* W1    = (const float*)d_in[13];
  const float* b1    = (const float*)d_in[14];
  const float* W2    = (const float*)d_in[15];
  const float* b2    = (const float*)d_in[16];
  const float* adW   = (const float*)d_in[17];
  const float* adb   = (const float*)d_in[18];
  const float* auW   = (const float*)d_in[19];
  const float* aub   = (const float*)d_in[20];
  const float* adgate= (const float*)d_in[21];
  const float* bng   = (const float*)d_in[22];
  const float* bnb   = (const float*)d_in[23];
  const float* bnm   = (const float*)d_in[24];
  const float* bnv   = (const float*)d_in[25];
  const float* ln3g  = (const float*)d_in[26];
  const float* ln3b  = (const float*)d_in[27];
  const float* aWqkv = (const float*)d_in[28];
  const float* aWproj= (const float*)d_in[29];
  const float* abproj= (const float*)d_in[30];
  const float* a1dW  = (const float*)d_in[31];
  const float* a1db  = (const float*)d_in[32];
  const float* a1uW  = (const float*)d_in[33];
  const float* a1ub  = (const float*)d_in[34];
  const int*   idx   = (const int*)d_in[35];
  const int*   cidx  = (const int*)d_in[36];
  const int*   ifmm  = (const int*)d_in[38];
  const int*   pcof  = (const int*)d_in[39];
  const int*   ccof  = (const int*)d_in[40];

  float* ws    = (float*)d_ws;
  float* xbuf  = ws + F_XBUF;
  float* xn    = ws + F_XN;
  float* qkvb  = ws + F_QKV;
  float* attnb = ws + F_ATTN;
  float* hbig  = ws + F_HBIG;
  float* hbuf  = ws + F_HBUF;
  float* admid = ws + F_ADMID;
  float* pqkvb = ws + F_PQKV;
  unsigned short* bw    = (unsigned short*)(ws + F_BW);
  unsigned short* bact  = (unsigned short*)(ws + F_BACT);
  unsigned short* bact2 = (unsigned short*)(ws + F_BACT2);
  float* out   = (float*)d_out;

  auto cvt = [&](const float* s, unsigned short* d, long n) {
    long n4 = n >> 2;
    cvt_bf16_kernel<<<(unsigned)((n4 + 255) / 256), 256, 0, stream>>>(s, d, n4);
  };
  auto gemm = [&](const unsigned short* A, const unsigned short* B, float* C,
                  int M, int N, int K, long sA, long sB, long sC, int batch,
                  const float* bias, int flags, float alpha, const int* aI) {
    dim3 grid((N + 127) / 128, (M + 63) / 64, batch);
    gemm_bf16<<<grid, 256, 0, stream>>>(A, B, C, M, N, K, sA, sB, sC, bias, flags, alpha, aI);
  };

  // ---- convert all weights (and prompt embed) to bf16 once ----
  cvt(Wqkv,  bw + oWQ,  384L * 1152);
  cvt(Wproj, bw + oWP,  384L * 384);
  cvt(W1,    bw + oW1,  384L * 1536);
  cvt(W2,    bw + oW2,  1536L * 384);
  cvt(adW,   bw + oAD,  384L * 16);
  cvt(auW,   bw + oAU,  16L * 384);
  cvt(aWqkv, bw + oA1Q, 384L * 1152);
  cvt(aWproj,bw + oA1P, 384L * 384);
  cvt(a1dW,  bw + oA1D, 384L * 16);
  cvt(a1uW,  bw + oA1U, 16L * 384);
  cvt(pe,    bw + oPE,  (long)PP * 384);

  // ---------------- phase 1: global block ----------------
  (void)hipMemcpyAsync(xbuf, x, (size_t)TOK * CH * sizeof(float),
                       hipMemcpyDeviceToDevice, stream);

  ln_kernel<<<TOK, 128, 0, stream>>>(x, ln1g, ln1b, xn, CH);
  gemm(bw + oPE, bw + oWQ, pqkvb, PP, LDQKV, CH, 0, 0, 0, 1, nullptr, 0, 1.0f, nullptr);
  cvt(xn, bact, (long)TOK * CH);
  gemm(bact, bw + oWQ, qkvb, TOK, LDQKV, CH, 0, 0, 0, 1, nullptr, 0, 1.0f, nullptr);

  attn_kernel<<<dim3(BATCH, NH, (T_TOK + 15) / 16), 256, 0, stream>>>(
      qkvb, attnb, T_TOK, T_TOK, T_TOK, CH, pqkvb, PP, gate, SCALE_ATT);

  cvt(attnb, bact, (long)TOK * CH);
  gemm(bact, bw + oWP, xbuf, TOK, CH, CH, 0, 0, 0, 1, bproj, GF_ACCUM, 1.0f, nullptr);

  ln_kernel<<<TOK, 128, 0, stream>>>(xbuf, ln2g, ln2b, xn, CH);
  cvt(xn, bact, (long)TOK * CH);
  gemm(bact, bw + oW1, hbig, TOK, HIDDEN, CH, 0, 0, 0, 1, b1, GF_GELU, 1.0f, nullptr);
  cvt(hbig, bact, (long)TOK * HIDDEN);
  gemm(bact, bw + oW2, hbuf, TOK, CH, HIDDEN, 0, 0, 0, 1, b2, 0, 1.0f, nullptr);
  cvt(hbuf, bact, (long)TOK * CH);
  gemm(bact, bw + oAD, admid, TOK, REDD, CH, 0, 0, 0, 1, adb, GF_GELU, 1.0f, nullptr);
  cvt(admid, bact2, (long)TOK * REDD);
  gemm(bact2, bw + oAU, hbuf, TOK, CH, REDD, 0, 0, 0, 1, aub, GF_ACCUM, 1.0f, nullptr);
  axpy_kernel<<<(unsigned)(((long)TOK * CH / 4 + 255) / 256), 256, 0, stream>>>(
      xbuf, hbuf, adgate, (long)TOK * CH / 4);

  // ---------------- phase 2: local neighborhood attention (chunked) ----------------
  float* xnb    = ws + F_XNB;
  float* lnnb   = ws + F_LNNB;
  float* qkvnb  = ws + F_QKVNB;
  float* attnnb = ws + F_ATTNNB;
  float* lcmax  = ws + F_LCMAX;
  float* lcmean = ws + F_LCMEAN;
  float* visb   = ws + F_VIS;
  float* wbuf   = ws + F_WBUF;

  for (int c = 0; c < NCHUNK; ++c) {
    const int rbase = c * CHUNK_ROWS;
    gather_rows_kernel<<<CHUNK_ROWS, 128, 0, stream>>>(xbuf, xnb, idx + rbase);
    ln_kernel<<<CHUNK_ROWS, 128, 0, stream>>>(xnb, ln3g, ln3b, lnnb, CH);
    cvt(lnnb, bact, (long)CHUNK_ROWS * CH);
    gemm(bact, bw + oA1Q, qkvnb, CHUNK_ROWS, LDQKV, CH, 0, 0, 0, 1, nullptr, 0, 1.0f, nullptr);
    attn_kernel<<<dim3(CHUNK_GRP, NH, 2), 256, 0, stream>>>(
        qkvnb, attnnb, GS_, GS_, GS_, CH, nullptr, 0, nullptr, SCALE_ATT);
    cvt(attnnb, bact, (long)CHUNK_ROWS * CH);
    gemm(bact, bw + oA1P, xnb, CHUNK_ROWS, CH, CH, 0, 0, 0, 1, abproj, GF_ACCUM, 1.0f, nullptr);
    group_reduce_kernel<<<CHUNK_GRP, 128, 0, stream>>>(xnb, lcmax, lcmean, CH, c * CHUNK_GRP);
  }

  // vis = gelu(BN(max[+mean])) + center_cof * x_ct
  {
    long total = (long)BATCH * S_NUM * CH;
    vis_kernel<<<(unsigned)((total + 255) / 256), 256, 0, stream>>>(
        lcmax, lcmean, bng, bnb, bnm, bnv, xbuf, cidx, ifmm, ccof, visb, total, CH);
  }
  // inverse-distance weights; xt += pro_cof * (w @ vis)
  wgt_kernel<<<BATCH * G_NUM, S_NUM, 0, stream>>>(c1, c2, wbuf);
  unsigned short* bvis = bact;                       // bf16 staging (phase-2 region free)
  unsigned short* bwgt = bact + 2000000L;
  cvt(visb, bvis, (long)BATCH * S_NUM * CH);
  cvt(wbuf, bwgt, (long)BATCH * G_NUM * S_NUM);
  gemm(bwgt, bvis, xbuf + CH /* skip cls row */, G_NUM, CH, S_NUM,
       (long)G_NUM * S_NUM, (long)S_NUM * CH, (long)T_TOK * CH, BATCH,
       nullptr, GF_ACCUM, 1.0f, pcof);

  // ---------------- final adapter: out = x + gelu(x@dW+db)@uW + ub ----------------
  (void)hipMemcpyAsync(out, xbuf, (size_t)TOK * CH * sizeof(float),
                       hipMemcpyDeviceToDevice, stream);
  cvt(xbuf, bact, (long)TOK * CH);
  gemm(bact, bw + oA1D, admid, TOK, REDD, CH, 0, 0, 0, 1, a1db, GF_GELU, 1.0f, nullptr);
  cvt(admid, bact2, (long)TOK * REDD);
  gemm(bact2, bw + oA1U, out, TOK, CH, REDD, 0, 0, 0, 1, a1ub, GF_ACCUM, 1.0f, nullptr);

  (void)in_sizes; (void)n_in; (void)out_size; (void)ws_size;
}
